// TextLSTM_9311489098010
// MI455X (gfx1250) — matrix operand order repaired and checked by reading, not compiled, not run on hardware
//
#include <hip/hip_runtime.h>

typedef __attribute__((ext_vector_type(16))) _Float16 v16h;
typedef __attribute__((ext_vector_type(8)))  _Float16 v8h;
typedef __attribute__((ext_vector_type(8)))  float    v8f;
typedef __attribute__((ext_vector_type(4)))  unsigned int u32x4;
typedef __attribute__((ext_vector_type(8)))  unsigned int u32x8;

#define N_CLASS 32000
#define EMB 256
#define NH  512
#define BB  128
#define TT  256
#define KW  (EMB + NH)   // 768
#define KSTEPS (KW / 32) // 24

union F8 { v8f v; float f[8]; };

// A-matrix 16x32 f16 fragment (contiguous K per lane, matching B):
// halves 0..15 -> K = h + 16*group.
__device__ __forceinline__ v16h load_fragA16(const _Float16* p, int group) {
    union { v16h v; v8h h[2]; } u;
    u.h[0] = *(const v8h*)(p + 16 * group);
    u.h[1] = *(const v8h*)(p + 16 * group + 8);
    return u.v;
}

// B-matrix 32x16 f16 fragment: lanes 0-15 hold K=0..15, lanes 16-31 hold
// K=16..31 (contiguous K per lane); weights stored [n][k] -> 16 contiguous
// halves at kk + 16*group.
__device__ __forceinline__ v16h load_fragB16(const _Float16* p, int group) {
    union { v16h v; v8h h[2]; } u;
    u.h[0] = *(const v8h*)(p + 16 * group);
    u.h[1] = *(const v8h*)(p + 16 * group + 8);
    return u.v;
}

__device__ __forceinline__ float sigmoidf_(float x) {
    return 1.0f / (1.0f + expf(-x));
}

#define WMMA(acc, a, b) \
    (acc) = __builtin_amdgcn_wmma_f32_16x16x32_f16(false, (a), false, (b), (short)0, (acc), false, false)

// --------- Tensor Data Mover: 2D tile (2-byte elements) global -> LDS ------
__device__ __forceinline__ void tdm_load_2d(unsigned lds_off, const void* gptr,
                                            unsigned dim0, unsigned dim1,
                                            unsigned tdim0, unsigned tdim1,
                                            unsigned stride0)
{
    const unsigned long long ga = (unsigned long long)(uintptr_t)gptr;
    u32x4 g0;
    g0.x = 1u;                                                // count=1, user-mode
    g0.y = lds_off;                                           // lds_addr (bytes)
    g0.z = (unsigned)(ga & 0xFFFFFFFFu);                      // global_addr[31:0]
    g0.w = (unsigned)((ga >> 32) & 0x1FFFFFFu) | (2u << 30);  // addr[56:32]|type=2
    u32x8 g1;
    g1.s0 = (1u << 16);                                  // wg_mask=0, data_size=2B
    g1.s1 = (dim0 & 0xFFFFu) << 16;                      // tensor_dim0[15:0]
    g1.s2 = ((dim0 >> 16) & 0xFFFFu) | ((dim1 & 0xFFFFu) << 16);
    g1.s3 = ((dim1 >> 16) & 0xFFFFu) | ((tdim0 & 0xFFFFu) << 16);
    g1.s4 = (tdim1 & 0xFFFFu);                           // tile_dim1, tile_dim2=0
    g1.s5 = stride0;                                     // tensor_dim0_stride[31:0]
    g1.s6 = 0u;
    g1.s7 = 0u;
    asm volatile("tensor_load_to_lds %0, %1" :: "s"(g0), "s"(g1) : "memory");
}

// ---------------- prep kernels ----------------

__global__ __launch_bounds__(256)
void embed_kernel(const int* __restrict__ X, const float* __restrict__ E,
                  _Float16* __restrict__ Xe)
{
    size_t idx = (size_t)blockIdx.x * blockDim.x + threadIdx.x;  // T*B*(EMB/4)
    const int e4 = (int)(idx & 63);
    const size_t rem = idx >> 6;
    const int b = (int)(rem & (BB - 1));
    const int t = (int)(rem >> 7);
    const int tok = X[(size_t)b * TT + t];
    const float4 v = *(const float4*)(E + (size_t)tok * EMB + e4 * 4);
    _Float16* d = Xe + (((size_t)t * BB + b) * EMB) + e4 * 4;
    d[0] = (_Float16)v.x; d[1] = (_Float16)v.y;
    d[2] = (_Float16)v.z; d[3] = (_Float16)v.w;
}

__global__ __launch_bounds__(256)
void build_wcat_kernel(const float* __restrict__ Wxf, const float* __restrict__ Whf,
                       const float* __restrict__ Wxi, const float* __restrict__ Whi,
                       const float* __restrict__ Wxc, const float* __restrict__ Whc,
                       const float* __restrict__ Wxo, const float* __restrict__ Who,
                       _Float16* __restrict__ Wcat)
{
    const size_t idx = (size_t)blockIdx.x * blockDim.x + threadIdx.x;  // 4*NH*KW
    const int k = (int)(idx % KW);
    const size_t rem = idx / KW;
    const int n = (int)(rem % NH);
    const int g = (int)(rem / NH);
    const float* px = (g == 0) ? Wxf : (g == 1) ? Wxi : (g == 2) ? Wxc : Wxo;
    const float* ph = (g == 0) ? Whf : (g == 1) ? Whi : (g == 2) ? Whc : Who;
    const float v = (k < EMB) ? px[(size_t)k * NH + n]
                              : ph[(size_t)(k - EMB) * NH + n];
    Wcat[idx] = (_Float16)v;
}

__global__ __launch_bounds__(256)
void build_wq_kernel(const float* __restrict__ Whq, _Float16* __restrict__ Wq)
{
    const size_t idx = (size_t)blockIdx.x * blockDim.x + threadIdx.x;  // N_CLASS*NH
    const int k = (int)(idx & (NH - 1));
    const size_t n = idx >> 9;
    Wq[idx] = (_Float16)Whq[(size_t)k * N_CLASS + n];
}

__global__ __launch_bounds__(256)
void init_state_kernel(const float* __restrict__ H, const float* __restrict__ C,
                       _Float16* __restrict__ H16, float* __restrict__ Cf)
{
    const int i = blockIdx.x * blockDim.x + threadIdx.x;  // BB*NH
    H16[i] = (_Float16)H[i];
    Cf[i]  = C[i];
}

// ---------------- fused recurrent step ----------------
// 32 blocks x 8 waves; all waves of a block share m_tile = blockIdx>>2, so the
// A tile ([Xe_t | Hin] rows, 24KB f16) is TDM-DMA'd to LDS once per block.
// Each wave owns one 16x16 (batch x hidden) tile for ALL FOUR gates; the K
// loop is software-pipelined one iteration ahead (prefetch A + 4 B fragments).
__global__ __launch_bounds__(256)
void lstm_step_kernel(const _Float16* __restrict__ XeT,   // [BB][EMB]
                      const _Float16* __restrict__ Wcat,  // [4][NH][KW] (n-major)
                      const _Float16* __restrict__ Hin,   // [BB][NH]
                      _Float16* __restrict__ Hout,        // [BB][NH]
                      float* __restrict__ Cst,            // [BB][NH] in-place
                      const float* __restrict__ bf, const float* __restrict__ bi,
                      const float* __restrict__ bc, const float* __restrict__ bo)
{
    __shared__ _Float16 AX[16 * EMB];   //  8 KB
    __shared__ _Float16 AH[16 * NH];    // 16 KB

    const int wave   = ((blockIdx.x * blockDim.x) + threadIdx.x) >> 5;
    const int lane   = threadIdx.x & 31;
    const int m_tile = wave >> 5;    // == blockIdx.x >> 2 (block-uniform)
    const int n_tile = wave & 31;
    const int group  = lane >> 4;
    const int l16    = lane & 15;

    if (threadIdx.x < 32) {
        const unsigned mrow0 = (unsigned)(blockIdx.x >> 2) * 16u;
        tdm_load_2d((unsigned)(uintptr_t)AX, XeT + (size_t)mrow0 * EMB,
                    EMB, 16, EMB, 16, EMB);
        tdm_load_2d((unsigned)(uintptr_t)AH, Hin + (size_t)mrow0 * NH,
                    NH, 16, NH, 16, NH);
        __builtin_amdgcn_s_wait_tensorcnt(0);
    }
    __syncthreads();

    const int brow = n_tile * 16 + l16;
    const _Float16* bF = Wcat + (size_t)(0 * NH + brow) * KW;
    const _Float16* bI = Wcat + (size_t)(1 * NH + brow) * KW;
    const _Float16* bC = Wcat + (size_t)(2 * NH + brow) * KW;
    const _Float16* bO = Wcat + (size_t)(3 * NH + brow) * KW;
    const _Float16* aX = AX + (size_t)l16 * EMB;
    const _Float16* aH = AH + (size_t)l16 * NH;

    // Issue epilogue operands early (independent of the GEMM).
    const int j = n_tile * 16 + l16;
    float c_old[8];
    #pragma unroll
    for (int r = 0; r < 8; ++r)
        c_old[r] = Cst[(size_t)(m_tile * 16 + r + 8 * group) * NH + j];
    const float bfv = bf[j], biv = bi[j], bcv = bc[j], bov = bo[j];

    F8 accF, accI, accC, accO;
    #pragma unroll
    for (int i = 0; i < 8; ++i) {
        accF.f[i] = 0.f; accI.f[i] = 0.f; accC.f[i] = 0.f; accO.f[i] = 0.f;
    }

    // A source per k-step (compile-time under full unroll): LDS X then LDS H.
    #define A_PTR(ks) (((ks) < EMB / 32) ? (aX + 32 * (ks)) : (aH + 32 * ((ks) - EMB / 32)))

    v16h a_c = load_fragA16(A_PTR(0), group);
    v16h f_c = load_fragB16(bF, group);
    v16h i_c = load_fragB16(bI, group);
    v16h c_c = load_fragB16(bC, group);
    v16h o_c = load_fragB16(bO, group);

    #pragma unroll
    for (int ks = 0; ks < KSTEPS; ++ks) {
        v16h a_n, f_n, i_n, c_n, o_n;
        if (ks + 1 < KSTEPS) {          // prefetch next iteration's fragments
            const int kk = (ks + 1) * 32;
            a_n = load_fragA16(A_PTR(ks + 1), group);
            f_n = load_fragB16(bF + kk, group);
            i_n = load_fragB16(bI + kk, group);
            c_n = load_fragB16(bC + kk, group);
            o_n = load_fragB16(bO + kk, group);
        }
        WMMA(accF.v, a_c, f_c);
        WMMA(accI.v, a_c, i_c);
        WMMA(accC.v, a_c, c_c);
        WMMA(accO.v, a_c, o_c);
        if (ks + 1 < KSTEPS) {
            a_c = a_n; f_c = f_n; i_c = i_n; c_c = c_n; o_c = o_n;
        }
    }
    #undef A_PTR

    // C/D layout: VGPR r, lanes 0-15 -> M=r, lanes 16-31 -> M=r+8; N = lane&15
    #pragma unroll
    for (int r = 0; r < 8; ++r) {
        const int bidx = m_tile * 16 + r + 8 * group;
        const size_t o = (size_t)bidx * NH + j;
        const float F = sigmoidf_(accF.f[r] + bfv);
        const float I = sigmoidf_(accI.f[r] + biv);
        const float G = tanhf(accC.f[r] + bcv);
        const float O = sigmoidf_(accO.f[r] + bov);
        const float h_new = O * tanhf(c_old[r]);   // uses OLD C, like reference
        Cst[o]  = F * c_old[r] + I * G;
        Hout[o] = (_Float16)h_new;
    }
}

// ---------------- final logits GEMM ----------------
__global__ __launch_bounds__(256)
void lstm_out_kernel(const _Float16* __restrict__ H,   // [BB][NH]
                     const _Float16* __restrict__ Wq,  // [N_CLASS][NH]
                     const float* __restrict__ bq,
                     float* __restrict__ out)          // [BB][N_CLASS]
{
    __shared__ _Float16 BW[16 * NH];   // 16 KB, shared by all 8 waves

    const int m_tile = threadIdx.x >> 5;   // 0..7
    const int n_tile = blockIdx.x;         // 0..1999
    const int lane   = threadIdx.x & 31;
    const int group  = lane >> 4;
    const int l16    = lane & 15;

    if (threadIdx.x < 32) {
        tdm_load_2d((unsigned)(uintptr_t)BW, Wq + (size_t)n_tile * 16 * NH,
                    16 * NH, 1, 16 * NH, 1, 16 * NH);  // contiguous 16KB tile
        __builtin_amdgcn_s_wait_tensorcnt(0);
    }
    __syncthreads();

    const _Float16* aH = H  + (size_t)(m_tile * 16 + l16) * NH;
    const _Float16* bW = BW + (size_t)l16 * NH;

    const int n = n_tile * 16 + l16;
    const float bqv = bq[n];

    F8 acc;
    #pragma unroll
    for (int i = 0; i < 8; ++i) acc.f[i] = 0.f;

    v16h a_c = load_fragA16(aH, group);
    v16h b_c = load_fragB16(bW, group);
    #pragma unroll
    for (int ks = 0; ks < NH / 32; ++ks) {
        v16h a_n, b_n;
        if (ks + 1 < NH / 32) {
            const int kk = (ks + 1) * 32;
            a_n = load_fragA16(aH + kk, group);
            b_n = load_fragB16(bW + kk, group);
        }
        WMMA(acc.v, a_c, b_c);
        if (ks + 1 < NH / 32) { a_c = a_n; b_c = b_n; }
    }

    #pragma unroll
    for (int r = 0; r < 8; ++r) {
        const int bidx = m_tile * 16 + r + 8 * group;
        out[(size_t)bidx * N_CLASS + n] = acc.f[r] + bqv;
    }
}

// ---------------- launch ----------------

extern "C" void kernel_launch(void* const* d_in, const int* in_sizes, int n_in,
                              void* d_out, int out_size, void* d_ws, size_t ws_size,
                              hipStream_t stream)
{
    (void)in_sizes; (void)n_in; (void)out_size; (void)ws_size;

    const int*   X   = (const int*)  d_in[0];
    const float* H   = (const float*)d_in[1];
    const float* C   = (const float*)d_in[2];
    const float* E   = (const float*)d_in[3];
    const float* Wxf = (const float*)d_in[4];
    const float* Whf = (const float*)d_in[5];
    const float* bf  = (const float*)d_in[6];
    const float* Wxi = (const float*)d_in[7];
    const float* Whi = (const float*)d_in[8];
    const float* bi  = (const float*)d_in[9];
    const float* Wxc = (const float*)d_in[10];
    const float* Whc = (const float*)d_in[11];
    const float* bc  = (const float*)d_in[12];
    const float* Wxo = (const float*)d_in[13];
    const float* Who = (const float*)d_in[14];
    const float* bo  = (const float*)d_in[15];
    const float* Whq = (const float*)d_in[16];
    const float* bq  = (const float*)d_in[17];
    float* out = (float*)d_out;

    // workspace layout (all slices 16B aligned)
    char* ws = (char*)d_ws;
    _Float16* Xe    = (_Float16*)(ws);                 // T*B*EMB f16      = 16 MB
    _Float16* Wcat  = (_Float16*)(ws + 16777216);      // 4*NH*KW f16      =  3 MB
    _Float16* Wq    = (_Float16*)(ws + 19922944);      // N_CLASS*NH f16   = 32 MB
    _Float16* Hbuf0 = (_Float16*)(ws + 52690944);      // BB*NH f16
    _Float16* Hbuf1 = (_Float16*)(ws + 52822016);      // BB*NH f16
    float*    Cst   = (float*)   (ws + 52953088);      // BB*NH f32

    embed_kernel<<<(TT * BB * (EMB / 4)) / 256, 256, 0, stream>>>(X, E, Xe);
    build_wcat_kernel<<<(4 * NH * KW) / 256, 256, 0, stream>>>(
        Wxf, Whf, Wxi, Whi, Wxc, Whc, Wxo, Who, Wcat);
    build_wq_kernel<<<(N_CLASS * NH) / 256, 256, 0, stream>>>(Whq, Wq);
    init_state_kernel<<<(BB * NH) / 256, 256, 0, stream>>>(H, C, Hbuf0, Cst);

    // 256 sequential fused steps, ping-pong H buffers (stream order = sync)
    for (int t = 0; t < TT; ++t) {
        const _Float16* Hin  = (t & 1) ? Hbuf1 : Hbuf0;
        _Float16*       Hout = (t & 1) ? Hbuf0 : Hbuf1;
        lstm_step_kernel<<<32, 256, 0, stream>>>(
            Xe + (size_t)t * BB * EMB, Wcat, Hin, Hout, Cst, bf, bi, bc, bo);
    }

    // T even -> final H landed back in Hbuf0
    lstm_out_kernel<<<N_CLASS / 16, 256, 0, stream>>>(Hbuf0, Wq, bq, out);
}
